// GCNBackbone_34394098107006
// MI455X (gfx1250) — matrix operand order repaired and checked by reading, not compiled, not run on hardware
//
#include <hip/hip_runtime.h>

#define N_HID 128
#define LN_EPS 1e-5f

typedef __attribute__((ext_vector_type(16))) __bf16 v16bf;
typedef __attribute__((ext_vector_type(8)))  __bf16 v8bf;
typedef __attribute__((ext_vector_type(4)))  __bf16 v4bf;
typedef __attribute__((ext_vector_type(8)))  float  v8f;
typedef __attribute__((ext_vector_type(4)))  float  v4f;

// ---------------- wave32 helpers ----------------
__device__ __forceinline__ float wave_sum32(float v) {
#pragma unroll
    for (int off = 16; off > 0; off >>= 1)
        v += __shfl_xor(v, off, 32);
    return v;
}

// ---------------- setup kernels ----------------
__global__ void k_zero(float* __restrict__ p, int n) {
    int i = blockIdx.x * blockDim.x + threadIdx.x;
    if (i < n) p[i] = 0.0f;
}

__global__ void k_deg(const long long* __restrict__ dst, const float* __restrict__ w,
                      float* __restrict__ deg, int E) {
    int e = blockIdx.x * blockDim.x + threadIdx.x;
    if (e < E) unsafeAtomicAdd(&deg[(int)dst[e]], w[e]);
}

__global__ void k_dinv(const float* __restrict__ deg, float* __restrict__ dinv,
                       float* __restrict__ nself, int N) {
    int i = blockIdx.x * blockDim.x + threadIdx.x;
    if (i < N) {
        float d = deg[i] + 1.0f;                 // + self-loop weight 1
        float r = (d > 0.0f) ? rsqrtf(d) : 0.0f;
        dinv[i]  = r;
        nself[i] = r * r;                        // self-loop edge norm
    }
}

__global__ void k_norm(const long long* __restrict__ src, const long long* __restrict__ dst,
                       const float* __restrict__ w, const float* __restrict__ dinv,
                       float* __restrict__ norm_e, int E) {
    int e = blockIdx.x * blockDim.x + threadIdx.x;
    if (e < E) norm_e[e] = dinv[(int)src[e]] * w[e] * dinv[(int)dst[e]];
}

// f32 -> bf16 elementwise
__global__ void k_cvt_bf16(const float* __restrict__ in, __bf16* __restrict__ out, int n) {
    int i = blockIdx.x * blockDim.x + threadIdx.x;
    if (i < n) out[i] = (__bf16)in[i];
}

// W[K][128] row-major f32  ->  Wt[128][K] col-major bf16
__global__ void k_transpose_cvt(const float* __restrict__ W, __bf16* __restrict__ Wt, int K) {
    int i = blockIdx.x * blockDim.x + threadIdx.x;
    if (i < K * N_HID) {
        int k = i >> 7;            // /128
        int c = i & (N_HID - 1);
        Wt[(size_t)c * K + k] = (__bf16)W[i];
    }
}

// ---------------- WMMA GEMM: C[M,128] = A[M,K] @ B[K,128] ----------------
// A: bf16 row-major [M,K].  Bt: bf16 col-major [128][K] (B transposed).
// Block = 16 rows x 128 cols, 128 threads = 4 waves, each wave owns two 16x16 tiles.
// Fragment packing used here (wave32, contiguous k order for A and B):
//   A 16x32 bf16 : lane -> M (mod 16); half j -> K = j + 16*hi
//   B 32x16 bf16 : lane -> N (mod 16); half j -> K = j + 16*hi
//   C 16x16 f32  : vgpr r -> M = r + 8*hi; lane -> N (mod 16)
// Epilogues:
//   bias/relu + optional bf16 shadow store (input projection)
//   optional fused agg init: agg = C * nself[row] + aggb[col] (layer GEMM)
template<int K>
__global__ __launch_bounds__(128)
void k_gemm_wmma(const __bf16* __restrict__ A, const __bf16* __restrict__ Bt,
                 const float* __restrict__ bias, float* __restrict__ C,
                 __bf16* __restrict__ Cbf, int relu,
                 const float* __restrict__ nself, const float* __restrict__ aggb,
                 float* __restrict__ agg) {
    const int lane = threadIdx.x & 31;
    const int wave = threadIdx.x >> 5;
    const int mloc = lane & 15;
    const int hi   = lane >> 4;           // 0: lanes 0-15, 1: lanes 16-31
    const int row0 = blockIdx.x << 4;     // 16 rows per block
    const int col0 = wave << 5;           // 32 cols per wave (two 16-wide tiles)

    v8f c0 = {};
    v8f c1 = {};

#pragma unroll
    for (int k0 = 0; k0 < K; k0 += 32) {
        // ---- A fragment: two contiguous 16B chunks per lane ----
        const __bf16* arow = A + (size_t)(row0 + mloc) * K + k0 + (hi << 4);
        v8bf alo = *(const v8bf*)(arow);        // K = k0 + 16*hi + [0,8)
        v8bf ahi = *(const v8bf*)(arow + 8);    // K = k0 + 16*hi + 8 + [0,8)
        v16bf a;
#pragma unroll
        for (int j = 0; j < 8; ++j) { a[j] = alo[j]; a[j + 8] = ahi[j]; }

        // ---- B fragments: 16 contiguous halves per lane (col-major weights) ----
        const __bf16* bc0 = Bt + (size_t)(col0 + mloc) * K + k0 + (hi << 4);
        const __bf16* bc1 = bc0 + (size_t)16 * K;
        v8bf b0lo = *(const v8bf*)(bc0);
        v8bf b0hi = *(const v8bf*)(bc0 + 8);
        v8bf b1lo = *(const v8bf*)(bc1);
        v8bf b1hi = *(const v8bf*)(bc1 + 8);
        v16bf b0, b1;
#pragma unroll
        for (int j = 0; j < 8; ++j) {
            b0[j] = b0lo[j]; b0[j + 8] = b0hi[j];
            b1[j] = b1lo[j]; b1[j + 8] = b1hi[j];
        }

        c0 = __builtin_amdgcn_wmma_f32_16x16x32_bf16(false, a, false, b0, (short)0, c0, false, false);
        c1 = __builtin_amdgcn_wmma_f32_16x16x32_bf16(false, a, false, b1, (short)0, c1, false, false);
    }

#pragma unroll
    for (int r = 0; r < 8; ++r) {
        int row = row0 + r + (hi << 3);
        int ca = col0 + mloc;
        int cb = ca + 16;
        float va = c0[r];
        float vb = c1[r];
        if (bias) { va += bias[ca]; vb += bias[cb]; }
        if (relu) { va = fmaxf(va, 0.0f); vb = fmaxf(vb, 0.0f); }
        size_t ia = (size_t)row * N_HID + ca;
        size_t ib = (size_t)row * N_HID + cb;
        C[ia] = va;
        C[ib] = vb;
        if (Cbf) { Cbf[ia] = (__bf16)va; Cbf[ib] = (__bf16)vb; }
        if (nself) {                              // fused self-loop + conv bias init
            float ns = nself[row];
            agg[ia] = va * ns + aggb[ca];
            agg[ib] = vb * ns + aggb[cb];
        }
    }
}

// ---------------- edge scatter ----------------
// one wave per edge: b128 gather of m[src] row, scale, f32 atomic-add into agg[dst] row
__global__ __launch_bounds__(256)
void k_scatter(const long long* __restrict__ src, const long long* __restrict__ dst,
               const float* __restrict__ norm_e, const float* __restrict__ m,
               float* __restrict__ agg, int E) {
    int lane = threadIdx.x & 31;
    int e = (blockIdx.x << 3) + (threadIdx.x >> 5);   // 8 waves/block
    if (e >= E) return;
    int s = (int)src[e];
    int d = (int)dst[e];
    float ne = norm_e[e];
    int c0 = lane << 2;                               // 4 contiguous channels/lane
    const float* mr = m + (size_t)s * N_HID + c0;
    float* ar = agg + (size_t)d * N_HID + c0;
    v4f mv = *(const v4f*)mr;
#pragma unroll
    for (int j = 0; j < 4; ++j)
        unsafeAtomicAdd(&ar[j], mv[j] * ne);
}

// ---------------- fused LayerNorm + ReLU + residual ----------------
// one wave per node, 4 contiguous channels per lane (b128 traffic);
// also refreshes the bf16 shadow of h for the next layer's WMMA GEMM
__global__ __launch_bounds__(256)
void k_ln_relu_res(const float* __restrict__ agg, const float* __restrict__ g,
                   const float* __restrict__ b, float* __restrict__ h,
                   __bf16* __restrict__ hbf, int N) {
    int lane = threadIdx.x & 31;
    int node = (blockIdx.x << 3) + (threadIdx.x >> 5);
    if (node >= N) return;
    int c0 = lane << 2;
    v4f v  = *(const v4f*)(agg + (size_t)node * N_HID + c0);
    float s  = v[0] + v[1] + v[2] + v[3];
    float s2 = v[0]*v[0] + v[1]*v[1] + v[2]*v[2] + v[3]*v[3];
    s  = wave_sum32(s);
    s2 = wave_sum32(s2);
    const float inv = 1.0f / (float)N_HID;
    float mu   = s * inv;
    float var  = s2 * inv - mu * mu;
    float rstd = rsqrtf(var + LN_EPS);
    v4f gv = *(const v4f*)(g + c0);
    v4f bv = *(const v4f*)(b + c0);
    float* hrow = h + (size_t)node * N_HID + c0;
    v4f hv = *(v4f*)hrow;
    v4bf bfv;
#pragma unroll
    for (int j = 0; j < 4; ++j) {
        float o  = (v[j] - mu) * rstd * gv[j] + bv[j];
        float nh = hv[j] + fmaxf(o, 0.0f);
        hv[j]  = nh;
        bfv[j] = (__bf16)nh;
    }
    *(v4f*)hrow = hv;
    *(v4bf*)(hbf + (size_t)node * N_HID + c0) = bfv;
}

// ---------------- driver ----------------
extern "C" void kernel_launch(void* const* d_in, const int* in_sizes, int n_in,
                              void* d_out, int out_size, void* d_ws, size_t ws_size,
                              hipStream_t stream) {
    const float*     x      = (const float*)d_in[0];
    const long long* eidx   = (const long long*)d_in[1];   // int64 [2,E]
    const float*     ew     = (const float*)d_in[2];
    const float*     W_in   = (const float*)d_in[3];
    const float*     b_in   = (const float*)d_in[4];
    const float*     conv_W = (const float*)d_in[5];
    const float*     conv_b = (const float*)d_in[6];
    const float*     ln_g   = (const float*)d_in[7];
    const float*     ln_b   = (const float*)d_in[8];

    const int IN_CH = 64;
    const int E = in_sizes[2];
    const int N = in_sizes[0] / IN_CH;

    const long long* srcI = eidx;
    const long long* dstI = eidx + E;

    // 256B-aligned workspace bump allocator
    char* ws = (char*)d_ws;
    size_t off = 0;
    auto alloc = [&](size_t bytes) -> char* {
        off = (off + 255) & ~(size_t)255;
        char* p = ws + off;
        off += bytes;
        return p;
    };

    float*  deg    = (float*)alloc((size_t)N * 4);
    float*  dinv   = (float*)alloc((size_t)N * 4);
    float*  nself  = (float*)alloc((size_t)N * 4);
    float*  norm_e = (float*)alloc((size_t)E * 4);
    float*  m      = (float*)alloc((size_t)N * N_HID * 4);
    float*  agg    = (float*)alloc((size_t)N * N_HID * 4);
    __bf16* x_bf   = (__bf16*)alloc((size_t)N * IN_CH * 2);
    __bf16* h_bf   = (__bf16*)alloc((size_t)N * N_HID * 2);
    __bf16* Wt_in  = (__bf16*)alloc((size_t)IN_CH * N_HID * 2);
    __bf16* Wt_c   = (__bf16*)alloc((size_t)4 * N_HID * N_HID * 2);
    float*  h      = (float*)d_out;           // N*128, lives in d_out

    // GCN normalization
    k_zero<<<(N + 255) / 256, 256, 0, stream>>>(deg, N);
    k_deg <<<(E + 255) / 256, 256, 0, stream>>>(dstI, ew, deg, E);
    k_dinv<<<(N + 255) / 256, 256, 0, stream>>>(deg, dinv, nself, N);
    k_norm<<<(E + 255) / 256, 256, 0, stream>>>(srcI, dstI, ew, dinv, norm_e, E);

    // bf16 operand prep
    k_cvt_bf16<<<(N * IN_CH + 255) / 256, 256, 0, stream>>>(x, x_bf, N * IN_CH);
    k_transpose_cvt<<<(IN_CH * N_HID + 255) / 256, 256, 0, stream>>>(W_in, Wt_in, IN_CH);
    for (int l = 0; l < 4; ++l)
        k_transpose_cvt<<<(N_HID * N_HID + 255) / 256, 256, 0, stream>>>(
            conv_W + (size_t)l * N_HID * N_HID, Wt_c + (size_t)l * N_HID * N_HID, N_HID);

    // input projection: h = relu(x @ W_in + b_in)    (N divisible by 16)
    k_gemm_wmma<64><<<N / 16, 128, 0, stream>>>(x_bf, Wt_in, b_in, h, h_bf, 1,
                                                nullptr, nullptr, nullptr);

    for (int l = 0; l < 4; ++l) {
        // m = h @ W_l ; fused: agg = m * nself + conv_b[l]
        k_gemm_wmma<128><<<N / 16, 128, 0, stream>>>(h_bf, Wt_c + (size_t)l * N_HID * N_HID,
                                                     nullptr, m, (__bf16*)nullptr, 0,
                                                     nself, conv_b + l * N_HID, agg);
        k_scatter<<<(E + 7) / 8, 256, 0, stream>>>(srcI, dstI, norm_e, m, agg, E);
        k_ln_relu_res<<<(N + 7) / 8, 256, 0, stream>>>(agg, ln_g + l * N_HID,
                                                       ln_b + l * N_HID, h, h_bf, N);
    }
}
